// LightweightAsymmetricAttention_14499809592014
// MI455X (gfx1250) — matrix operand order repaired and checked by reading, not compiled, not run on hardware
//
#include <hip/hip_runtime.h>
#include <hip/hip_bf16.h>

// ---------------------------------------------------------------------------
// Types / WMMA helper
// ---------------------------------------------------------------------------
typedef __attribute__((ext_vector_type(16))) __bf16 bf16x16;
typedef __attribute__((ext_vector_type(8)))  __bf16 bf16x8;
typedef __attribute__((ext_vector_type(8)))  float  f32x8;
typedef __attribute__((ext_vector_type(4)))  unsigned int u32x4;
typedef __attribute__((ext_vector_type(8)))  int i32x8t;
typedef __attribute__((ext_vector_type(4)))  int i32x4t;

__device__ __forceinline__ f32x8 wmma_bf16(bf16x16 a, bf16x16 b, f32x8 c) {
  return __builtin_amdgcn_wmma_f32_16x16x32_bf16(false, a, false, b,
                                                 (short)0, c, false, false);
}
__device__ __forceinline__ bf16x16 join8(bf16x8 lo, bf16x8 hi) {
  return __builtin_shufflevector(lo, hi, 0, 1, 2, 3, 4, 5, 6, 7,
                                 8, 9, 10, 11, 12, 13, 14, 15);
}

// TDM: DMA a 2-D tile (tile_w x tile_h elements of 2 bytes, row stride
// stride_elems) from global memory into LDS at byte offset lds_off.
// Descriptor bit layout per CDNA5 ISA sec. 8.3/8.4 (D# groups 0/1; 2-D tile).
// This toolchain exposes the 6-arg builtin: (g0, g1, g2, g3, g4, cpol).
__device__ __forceinline__ void tdm_load_2d(unsigned lds_off, const void* gptr,
                                            unsigned tile_w, unsigned tile_h,
                                            unsigned stride_elems) {
  unsigned long long ga = (unsigned long long)(uintptr_t)gptr;
  u32x4 g0;
  g0[0] = 1u;                                    // count=1 (valid), user D#
  g0[1] = lds_off;                               // LDS byte address
  g0[2] = (unsigned)ga;                          // global_addr[31:0]
  g0[3] = (unsigned)((ga >> 32) & 0x1FFFFFFull)  // global_addr[56:32]
          | (2u << 30);                          // type = 2 ("image")
  unsigned dim0 = stride_elems;                  // >= tile_w -> no OOB clip
  unsigned dim1 = 1u << 20;
  unsigned long long s0 = (unsigned long long)stride_elems;
  i32x8t g1;
  g1[0] = (int)(1u << 16);                                   // data_size = 2B
  g1[1] = (int)((dim0 & 0xFFFFu) << 16);                     // dim0[15:0]
  g1[2] = (int)((dim0 >> 16) | ((dim1 & 0xFFFFu) << 16));    // dim0 hi, dim1 lo
  g1[3] = (int)((dim1 >> 16) | (tile_w << 16));              // dim1 hi, tile0
  g1[4] = (int)tile_h;                                       // tile1, tile2=0
  g1[5] = (int)(unsigned)(s0 & 0xFFFFFFFFull);               // stride0[31:0]
  g1[6] = (int)(unsigned)(s0 >> 32);                         // stride0[47:32]
  g1[7] = 0;
  i32x4t gz4 = {0, 0, 0, 0};
  i32x8t gz8 = {0, 0, 0, 0, 0, 0, 0, 0};
  __builtin_amdgcn_tensor_load_to_lds(g0, g1, gz4, gz4, gz8, 0);
}

// Problem constants
#define BB   4
#define CC   512
#define HH   64
#define WW   64
#define NH_  8
#define DK_  64
#define NN   4096      // H*W
#define NR   1024      // (H/2)*(W/2)
#define KKV  2048      // C*R*R
#define MQ   16384     // B*N
#define MKV  4096      // B*Nr

// ---------------------------------------------------------------------------
// fp32 -> bf16 elementwise copy (weights are already B^T = [N][K] layout)
// ---------------------------------------------------------------------------
__global__ __launch_bounds__(256) void cvt_bf16(const float* __restrict__ src,
                                                __bf16* __restrict__ dst,
                                                int n) {
  int i = blockIdx.x * 256 + threadIdx.x;
  if (i < n) dst[i] = (__bf16)src[i];
}

// ---------------------------------------------------------------------------
// Activation packing: fp32 NCHW -> bf16 GEMM row-major layouts
// ---------------------------------------------------------------------------
__global__ __launch_bounds__(256) void pack_xq(const float* __restrict__ x,
                                               __bf16* __restrict__ Xq) {
  size_t i = (size_t)blockIdx.x * 256 + threadIdx.x;   // over MQ*CC
  int c = (int)(i & (CC - 1));
  size_t row = i >> 9;                 // b*N + n
  int b = (int)(row >> 12);
  int n = (int)(row & (NN - 1));
  Xq[i] = (__bf16)x[((size_t)(b * CC + c)) * NN + n];
}

__global__ __launch_bounds__(256) void pack_xkv(const float* __restrict__ x,
                                                __bf16* __restrict__ Xkv) {
  size_t i = (size_t)blockIdx.x * 256 + threadIdx.x;   // over MKV*KKV
  int colI = (int)(i & (KKV - 1));       // c*4 + kh*2 + kw
  size_t row = i >> 11;                  // b*Nr + p
  int b  = (int)(row >> 10);
  int p  = (int)(row & (NR - 1));
  int c  = colI >> 2;
  int kh = (colI >> 1) & 1;
  int kw = colI & 1;
  int h2 = p >> 5, w2 = p & 31;
  Xkv[i] = (__bf16)x[((size_t)(b * CC + c)) * NN + (h2 * 2 + kh) * WW + w2 * 2 + kw];
}

// V head-transpose: VbT[((b*8+h)*64+dk)*1024 + key] = Vb[(b*Nr+key)*512 + h*64+dk]
__global__ __launch_bounds__(256) void transpose_v(const __bf16* __restrict__ Vb,
                                                   __bf16* __restrict__ VbT) {
  size_t i = (size_t)blockIdx.x * 256 + threadIdx.x;   // over BB*NH_*DK_*NR
  int key = (int)(i & (NR - 1));
  size_t t = i >> 10;
  int dk = (int)(t & 63);
  size_t bh = t >> 6;                  // b*8 + h
  int b = (int)(bh >> 3), h = (int)(bh & 7);
  VbT[i] = Vb[((size_t)b * NR + key) * CC + h * DK_ + dk];
}

// ---------------------------------------------------------------------------
// bf16 WMMA GEMM: C[M,N] = A[M,K] * Bt[N,K]^T  (Bt is B transposed, row-major)
// 64x64 block tile, 128 threads (4 waves); wave w -> rows [w*16, w*16+16).
// Tile staging via Tensor Data Mover (double-buffered, TENSORcnt-pipelined);
// fragment builds are 16B/32B vector LDS loads.
// ---------------------------------------------------------------------------
__global__ __launch_bounds__(128)
void gemm_bf16(const __bf16* __restrict__ A, const __bf16* __restrict__ Bt,
               float* __restrict__ Cf, __bf16* __restrict__ Cb,
               int M, int N, int K, float out_scale) {
  __shared__ __bf16 As[2][2048];   // [buf][m*32 + k] : 64x32 tiles
  __shared__ __bf16 Bs[2][2048];   // [buf][n*32 + k]

  const int tid  = threadIdx.x;
  const int wave = tid >> 5;
  const int lane = tid & 31;
  const int col  = lane & 15;
  const int hi   = lane >> 4;
  const int m0   = blockIdx.x * 64;
  const int n0   = blockIdx.y * 64;

  const unsigned asOff = (unsigned)(uintptr_t)&As[0][0];
  const unsigned bsOff = (unsigned)(uintptr_t)&Bs[0][0];

  f32x8 acc[4];
#pragma unroll
  for (int nt = 0; nt < 4; ++nt)
#pragma unroll
    for (int j = 0; j < 8; ++j) acc[nt][j] = 0.0f;

  // Preload first tile pair into buffer 0 (TDM, wave 0 issues for the block)
  if (wave == 0) {
    tdm_load_2d(asOff, A + (size_t)m0 * K, 32, 64, (unsigned)K);
    tdm_load_2d(bsOff, Bt + (size_t)n0 * K, 32, 64, (unsigned)K);
  }

  const int nk = K / 32;
  for (int it = 0; it < nk; ++it) {
    const int buf = it & 1;
    if (wave == 0) {
      if (it + 1 < nk) {   // issue next pair, then wait for current pair
        const int k1 = (it + 1) * 32;
        tdm_load_2d(asOff + (buf ^ 1) * 4096, A + (size_t)m0 * K + k1,
                    32, 64, (unsigned)K);
        tdm_load_2d(bsOff + (buf ^ 1) * 4096, Bt + (size_t)n0 * K + k1,
                    32, 64, (unsigned)K);
        __builtin_amdgcn_s_wait_tensorcnt(2);
      } else {
        __builtin_amdgcn_s_wait_tensorcnt(0);
      }
    }
    __syncthreads();       // current buffer ready for all waves

    const __bf16* at = &As[buf][0];
    const __bf16* bt = &Bs[buf][0];

    // A fragment: two 16B runs forming one contiguous 32B k run (k = 16*hi + s)
    bf16x8 alo = *(const bf16x8*)(at + (wave * 16 + col) * 32 + hi * 16);
    bf16x8 ahi = *(const bf16x8*)(at + (wave * 16 + col) * 32 + hi * 16 + 8);
    bf16x16 af = join8(alo, ahi);

#pragma unroll
    for (int nt = 0; nt < 4; ++nt) {
      // B fragment: one contiguous 32B run (k = 16*hi + s, same order as A)
      bf16x16 bfv = *(const bf16x16*)(bt + (nt * 16 + col) * 32 + hi * 16);
      acc[nt] = wmma_bf16(af, bfv, acc[nt]);
    }
    __syncthreads();       // readers done before TDM overwrites this buffer
  }

#pragma unroll
  for (int nt = 0; nt < 4; ++nt) {
#pragma unroll
    for (int j = 0; j < 8; ++j) {
      int rr = m0 + wave * 16 + j + (hi ? 8 : 0);
      int cc2 = n0 + nt * 16 + col;
      float v = acc[nt][j] * out_scale;
      if (Cb) Cb[(size_t)rr * N + cc2] = (__bf16)v;
      else    Cf[(size_t)rr * N + cc2] = v;
    }
  }
}

// ---------------------------------------------------------------------------
// Attention: one wave per (b, head, 16-query tile).
// Qb/Kb: bf16 [rows][512], col = h*64+dk (Qb pre-scaled by dk^-0.5).
// VbT:   bf16 [(b*8+h)*64+dk][1024 keys]  (PV B-frags = 32B global loads).
// Dynamic LDS: sc fp32 [16][1032] scores + probs bf16 [16][1040].
// ---------------------------------------------------------------------------
#define SCS 1032   // score row stride (floats)
#define PRS 1040   // prob row stride (halfs)

__global__ __launch_bounds__(32)
void attn_kernel(const __bf16* __restrict__ Qb, const __bf16* __restrict__ Kb,
                 const __bf16* __restrict__ VbT, const float* __restrict__ bias,
                 __bf16* __restrict__ Aout) {
  extern __shared__ __attribute__((aligned(128))) char smem[];
  float*  sc    = (float*)smem;                        // [16][SCS]
  __bf16* probs = (__bf16*)(smem + 16 * SCS * 4);      // [16][PRS]

  const int lane = threadIdx.x;
  const int col  = lane & 15;
  const int hi   = lane >> 4;
  const int bid  = blockIdx.x;
  const int qt = bid & 255;
  const int h  = (bid >> 8) & 7;
  const int b  = bid >> 11;

  const size_t qrow0 = (size_t)b * NN + qt * 16;
  const size_t krow0 = (size_t)b * NR;
  const int hoff = h * DK_;
  const size_t vtb = ((size_t)(b * NH_ + h)) * DK_ * NR;

  // Q A-fragments (k = dk 0..31 and 32..63): 16B vector loads
  const __bf16* qp = Qb + (qrow0 + col) * CC + hoff;
  bf16x16 aq0 = join8(*(const bf16x8*)(qp + hi * 16),
                      *(const bf16x8*)(qp + hi * 16 + 8));
  bf16x16 aq1 = join8(*(const bf16x8*)(qp + 32 + hi * 16),
                      *(const bf16x8*)(qp + 40 + hi * 16));
  const float bv = bias[h];

  // ---- scores: Q (16x64) x K^T (64 x Nr), 16 keys per tile ----
  for (int kt = 0; kt < 64; ++kt) {
    const __bf16* kp = Kb + (krow0 + kt * 16 + col) * CC + hoff + hi * 16;
    bf16x16 bk0 = *(const bf16x16*)(kp);        // dk 0..31 half
    bf16x16 bk1 = *(const bf16x16*)(kp + 32);   // dk 32..63 half
    f32x8 acc;
#pragma unroll
    for (int j = 0; j < 8; ++j) acc[j] = bv;    // + attention bias
    acc = wmma_bf16(aq0, bk0, acc);
    acc = wmma_bf16(aq1, bk1, acc);
#pragma unroll
    for (int j = 0; j < 8; ++j)
      sc[(j + (hi ? 8 : 0)) * SCS + kt * 16 + col] = acc[j];
  }
  __syncthreads();

  // ---- softmax over 1024 keys: 2 lanes per row, float4-vectorized ----
  {
    float*  rowp = sc + col * SCS + hi * 512;
    __bf16* prow = probs + col * PRS + hi * 512;
    const float4* r4 = (const float4*)rowp;
    float mx = -3.4e38f;
    for (int i = 0; i < 128; ++i) {
      float4 v = r4[i];
      mx = fmaxf(mx, fmaxf(fmaxf(v.x, v.y), fmaxf(v.z, v.w)));
    }
    mx = fmaxf(mx, __shfl_xor(mx, 16));
    float s = 0.0f;
    for (int i = 0; i < 128; ++i) {
      float4 v = r4[i];
      float4 e;
      e.x = __expf(v.x - mx); e.y = __expf(v.y - mx);
      e.z = __expf(v.z - mx); e.w = __expf(v.w - mx);
      s += (e.x + e.y) + (e.z + e.w);
      ((float4*)rowp)[i] = e;
    }
    s += __shfl_xor(s, 16);
    float r = 1.0f / s;
    for (int i = 0; i < 64; ++i) {
      float4 a = r4[2 * i], b4 = r4[2 * i + 1];
      bf16x8 pv;
      pv[0] = (__bf16)(a.x * r);  pv[1] = (__bf16)(a.y * r);
      pv[2] = (__bf16)(a.z * r);  pv[3] = (__bf16)(a.w * r);
      pv[4] = (__bf16)(b4.x * r); pv[5] = (__bf16)(b4.y * r);
      pv[6] = (__bf16)(b4.z * r); pv[7] = (__bf16)(b4.w * r);
      *(bf16x8*)(prow + i * 8) = pv;
    }
  }
  __syncthreads();

  // ---- out = attn (16x1024) x V (1024x64) ----
  f32x8 ao[4];
#pragma unroll
  for (int nt = 0; nt < 4; ++nt)
#pragma unroll
    for (int j = 0; j < 8; ++j) ao[nt][j] = 0.0f;

  for (int kc = 0; kc < 32; ++kc) {
    const __bf16* pr = probs + col * PRS + kc * 32;
    bf16x16 af = join8(*(const bf16x8*)(pr + hi * 16),
                       *(const bf16x8*)(pr + hi * 16 + 8));
#pragma unroll
    for (int nt = 0; nt < 4; ++nt) {
      bf16x16 bfv = *(const bf16x16*)(VbT + vtb +
                                      (size_t)(nt * 16 + col) * NR +
                                      kc * 32 + hi * 16);
      ao[nt] = wmma_bf16(af, bfv, ao[nt]);
    }
  }

#pragma unroll
  for (int nt = 0; nt < 4; ++nt)
#pragma unroll
    for (int j = 0; j < 8; ++j) {
      size_t rr = qrow0 + j + (hi ? 8 : 0);
      Aout[rr * CC + hoff + nt * 16 + col] = (__bf16)ao[nt][j];
    }
}

// ---------------------------------------------------------------------------
// BatchNorm batch statistics: 16 channels per block, coalesced reads.
// ---------------------------------------------------------------------------
__global__ __launch_bounds__(256)
void bn_stats(const float* __restrict__ P, float* __restrict__ stats) {
  __shared__ float s1[256], s2[256];
  const int c0 = blockIdx.x * 16;
  const int tid = threadIdx.x;
  const int c = c0 + (tid & 15);
  float sum = 0.0f, sq = 0.0f;
  for (int r = tid >> 4; r < MQ; r += 16) {
    float v = P[(size_t)r * CC + c];
    sum += v; sq += v * v;
  }
  s1[tid] = sum; s2[tid] = sq;
  __syncthreads();
  for (int s = 128; s >= 16; s >>= 1) {
    if (tid < s) { s1[tid] += s1[tid + s]; s2[tid] += s2[tid + s]; }
    __syncthreads();
  }
  if (tid < 16) {
    float mu = s1[tid] * (1.0f / MQ);
    stats[c0 + tid] = mu;
    stats[CC + c0 + tid] = s2[tid] * (1.0f / MQ) - mu * mu;
  }
}

// ---------------------------------------------------------------------------
// Fused: BN-apply + residual + channel LayerNorm. One block per (b, n).
// ---------------------------------------------------------------------------
__global__ __launch_bounds__(256)
void fuse_final(const float* __restrict__ x, const float* __restrict__ P,
                const float* __restrict__ stats,
                const float* __restrict__ bng, const float* __restrict__ bnb,
                const float* __restrict__ lng, const float* __restrict__ lnb,
                float* __restrict__ out) {
  __shared__ float s1[256], s2[256];
  const int pos = blockIdx.x;        // b*4096 + n
  const int b = pos >> 12, n = pos & (NN - 1);
  const int tid = threadIdx.x;

  float y[2], sum = 0.0f, sq = 0.0f;
#pragma unroll
  for (int it = 0; it < 2; ++it) {
    int c = tid + it * 256;
    float p = P[(size_t)pos * CC + c];
    float bn = (p - stats[c]) * rsqrtf(stats[CC + c] + 1e-5f) * bng[c] + bnb[c];
    float v = x[((size_t)(b * CC + c)) * NN + n] + bn;
    y[it] = v; sum += v; sq += v * v;
  }
  s1[tid] = sum; s2[tid] = sq;
  __syncthreads();
  for (int s = 128; s > 0; s >>= 1) {
    if (tid < s) { s1[tid] += s1[tid + s]; s2[tid] += s2[tid + s]; }
    __syncthreads();
  }
  float mu = s1[0] * (1.0f / CC);
  float var = s2[0] * (1.0f / CC) - mu * mu;
  float r = rsqrtf(var + 1e-5f);
#pragma unroll
  for (int it = 0; it < 2; ++it) {
    int c = tid + it * 256;
    out[((size_t)(b * CC + c)) * NN + n] = (y[it] - mu) * r * lng[c] + lnb[c];
  }
}

// ---------------------------------------------------------------------------
// Host launcher
// ---------------------------------------------------------------------------
extern "C" void kernel_launch(void* const* d_in, const int* in_sizes, int n_in,
                              void* d_out, int out_size, void* d_ws, size_t ws_size,
                              hipStream_t stream) {
  (void)in_sizes; (void)n_in; (void)out_size; (void)ws_size;

  const float* x    = (const float*)d_in[0];
  const float* Wq   = (const float*)d_in[1];
  const float* Wk   = (const float*)d_in[2];
  const float* Wv   = (const float*)d_in[3];
  const float* bias = (const float*)d_in[4];
  const float* Wp   = (const float*)d_in[5];
  const float* bng  = (const float*)d_in[6];
  const float* bnb  = (const float*)d_in[7];
  const float* lng  = (const float*)d_in[8];
  const float* lnb  = (const float*)d_in[9];
  float* out = (float*)d_out;

  // workspace layout (bytes)
  char* ws = (char*)d_ws;
  size_t off = 0;
  __bf16* Xq   = (__bf16*)(ws + off); off += (size_t)MQ  * CC  * 2;  // 16 MB
  __bf16* Xkv  = (__bf16*)(ws + off); off += (size_t)MKV * KKV * 2;  // 16 MB
  __bf16* Btq  = (__bf16*)(ws + off); off += (size_t)CC  * CC  * 2;  // B^T = W layouts
  __bf16* Btk  = (__bf16*)(ws + off); off += (size_t)CC  * KKV * 2;
  __bf16* Btv  = (__bf16*)(ws + off); off += (size_t)CC  * KKV * 2;
  __bf16* Btp  = (__bf16*)(ws + off); off += (size_t)CC  * CC  * 2;
  __bf16* Qb   = (__bf16*)(ws + off); off += (size_t)MQ  * CC  * 2;  // 16 MB
  __bf16* Kb   = (__bf16*)(ws + off); off += (size_t)MKV * CC  * 2;  //  4 MB
  __bf16* Vb   = (__bf16*)(ws + off); off += (size_t)MKV * CC  * 2;  //  4 MB
  __bf16* VbT  = (__bf16*)(ws + off); off += (size_t)MKV * CC  * 2;  //  4 MB
  __bf16* Aout = (__bf16*)(ws + off); off += (size_t)MQ  * CC  * 2;  // 16 MB
  float*  P    = (float*)(ws + off);  off += (size_t)MQ  * CC  * 4;  // 32 MB
  float*  stats= (float*)(ws + off);  off += 2 * CC * 4;

  // 1. packing (weights: pure bf16 casts — they are already B^T layout)
  pack_xq <<<(MQ  * CC) / 256, 256, 0, stream>>>(x, Xq);
  pack_xkv<<<(MKV * KKV) / 256, 256, 0, stream>>>(x, Xkv);
  cvt_bf16<<<(CC * CC)  / 256, 256, 0, stream>>>(Wq, Btq, CC * CC);
  cvt_bf16<<<(CC * KKV) / 256, 256, 0, stream>>>(Wk, Btk, CC * KKV);
  cvt_bf16<<<(CC * KKV) / 256, 256, 0, stream>>>(Wv, Btv, CC * KKV);
  cvt_bf16<<<(CC * CC)  / 256, 256, 0, stream>>>(Wp, Btp, CC * CC);

  // 2. projections (bf16 WMMA GEMMs, TDM-staged); Q pre-scaled by dk^-0.5
  gemm_bf16<<<dim3(MQ / 64,  CC / 64), 128, 0, stream>>>(Xq,  Btq, nullptr, Qb,
                                                         MQ,  CC, CC,  0.125f);
  gemm_bf16<<<dim3(MKV / 64, CC / 64), 128, 0, stream>>>(Xkv, Btk, nullptr, Kb,
                                                         MKV, CC, KKV, 1.0f);
  gemm_bf16<<<dim3(MKV / 64, CC / 64), 128, 0, stream>>>(Xkv, Btv, nullptr, Vb,
                                                         MKV, CC, KKV, 1.0f);
  transpose_v<<<(MKV * CC) / 256, 256, 0, stream>>>(Vb, VbT);

  // 3. attention (dynamic LDS: 16*1032*4 + 16*1040*2 = 99328 B)
  attn_kernel<<<BB * NH_ * (NN / 16), 32, 99328, stream>>>(Qb, Kb, VbT, bias, Aout);

  // 4. proj GEMM (fp32 out)
  gemm_bf16<<<dim3(MQ / 64, CC / 64), 128, 0, stream>>>(Aout, Btp, P, nullptr,
                                                        MQ, CC, CC, 1.0f);

  // 5. BN stats + fused BN/residual/LayerNorm
  bn_stats  <<<CC / 16, 256, 0, stream>>>(P, stats);
  fuse_final<<<MQ, 256, 0, stream>>>(x, P, stats, bng, bnb, lng, lnb, out);
}
